// GINModule_33328946217389
// MI455X (gfx1250) — matrix operand order repaired and checked by reading, not compiled, not run on hardware
//
#include <hip/hip_runtime.h>
#include <hip/hip_bf16.h>

// ---------------------------------------------------------------------------
// GIN (3 layers, C=H=O=128) for MI455X / gfx1250.
//  * GEMMs: V_WMMA_F32_16X16X32_BF16, f32 accumulate, fused MLP per layer.
//  * Aggregation: zeroed f32 buffer + native global_atomic_add_f32 scatter;
//    "x + agg" fused into the WMMA kernel's bf16 LDS staging.
//  * Inter-layer activations carried as bf16 (halves gather traffic).
//  * Mean-pool accumulation fused into the GEMM-2 epilogue (pre-rounding f32).
// Workspace: agg f32[N*128] | h0 bf16[N*128] | h1 bf16[N*128]
//            | wpack bf16[6*16384] | ab f32[3*256] | pooled f32[512*384]
//            | cnt f32[512]
// ---------------------------------------------------------------------------

typedef __bf16 v16bf __attribute__((ext_vector_type(16)));
typedef __bf16 v8bf  __attribute__((ext_vector_type(8)));
typedef __bf16 v4bf  __attribute__((ext_vector_type(4)));
typedef float  v8f   __attribute__((ext_vector_type(8)));

#define LDS_STRIDE 136   // 128 + 8 pad -> row delta = 68 dwords -> rotates banks
#define GDIM       512   // graphs in batch

__device__ __forceinline__ void atomAddF32(float* p, float v) {
  unsafeAtomicAdd(p, v);   // native global_atomic_add_f32 (no CAS loop)
}

__device__ __forceinline__ v8f wmma_bf16(v16bf a, v16bf b, v8f c) {
  return __builtin_amdgcn_wmma_f32_16x16x32_bf16(
      /*neg_a=*/false, a, /*neg_b=*/false, b,
      /*c_mod=*/(short)0, c, /*reuse_a=*/false, /*reuse_b=*/false);
}

// A fragment (16x32 bf16) from LDS, ISA layout:
//  lanes 0-15 : M=lane,    elems 0..7 = K kb..kb+7, 8..15 = K kb+16..kb+23
//  lanes16-31 : M=lane-16, same but kb += 8
__device__ __forceinline__ v16bf load_afrag(const __bf16* lds, int waveRow,
                                            int lane, int kt) {
  const int row = waveRow + (lane & 15);
  const int kb  = kt * 32 + ((lane & 16) ? 8 : 0);
  union { v16bf v; v8bf h[2]; } u;
  u.h[0] = *(const v8bf*)(lds + row * LDS_STRIDE + kb);
  u.h[1] = *(const v8bf*)(lds + row * LDS_STRIDE + kb + 16);
  return u.v;
}

// B fragment (32x16 bf16): pre-packed so a lane's 16 values are contiguous.
__device__ __forceinline__ v16bf load_bfrag(const __bf16* __restrict__ wp,
                                            int lane, int kt, int nt) {
  const __bf16* p = wp + (((kt * 8 + nt) * 32 + lane) << 4);
  union { v16bf v; v8bf h[2]; } u;
  u.h[0] = *(const v8bf*)(p);
  u.h[1] = *(const v8bf*)(p + 8);
  return u.v;
}

// ---- weight packing: f32 [K=128][N=128] -> bf16 B-fragment layout ----------
__global__ __launch_bounds__(256) void pack_w(
    const float* __restrict__ w0, const float* __restrict__ w1,
    const float* __restrict__ w2, const float* __restrict__ w3,
    const float* __restrict__ w4, const float* __restrict__ w5,
    __bf16* __restrict__ dst) {
  int tid = blockIdx.x * 256 + threadIdx.x;
  if (tid >= 6 * 16384) return;
  const int m    = tid >> 14;
  const int o    = tid & 16383;
  const int j    = o & 15;
  const int lane = (o >> 4) & 31;
  const int nt   = (o >> 9) & 7;
  const int kt   = o >> 12;
  const int K = kt * 32 + ((lane & 16) ? 8 : 0) + (j < 8 ? j : j + 8);
  const int n = nt * 16 + (lane & 15);
  const float* src = (m == 0) ? w0 : (m == 1) ? w1 : (m == 2) ? w2
                   : (m == 3) ? w3 : (m == 4) ? w4 : w5;
  dst[tid] = (__bf16)src[K * 128 + n];
}

// ---- fold Linear bias + BN(eval) into alpha/beta ---------------------------
__global__ __launch_bounds__(128) void fold_bn(
    const float* __restrict__ b1, const float* __restrict__ gm,
    const float* __restrict__ bt, const float* __restrict__ rm,
    const float* __restrict__ rv, float* __restrict__ alpha,
    float* __restrict__ beta) {
  int c = threadIdx.x;
  float a = gm[c] * rsqrtf(rv[c] + 1e-5f);
  alpha[c] = a;
  beta[c]  = (b1[c] - rm[c]) * a + bt[c];
}

__global__ __launch_bounds__(256) void zero_f32(float* __restrict__ p, int n) {
  int i = blockIdx.x * 256 + threadIdx.x;
  if (i < n) p[i] = 0.0f;
}

// ---- edge scatter: agg[dst] += x_in[src]; one wave32 per edge --------------
__global__ __launch_bounds__(256) void edge_scatter_f32(
    float* __restrict__ agg, const float* __restrict__ xin,
    const int* __restrict__ ei, int nEdges) {
  const int lane = threadIdx.x & 31;
  const int e    = blockIdx.x * 8 + (threadIdx.x >> 5);
  if (e >= nEdges) return;
  const int s = ei[e];
  const int d = ei[nEdges + e];
  float4 v = *(const float4*)(xin + (size_t)s * 128 + lane * 4);
  float* p = agg + (size_t)d * 128 + lane * 4;
  atomAddF32(p + 0, v.x);
  atomAddF32(p + 1, v.y);
  atomAddF32(p + 2, v.z);
  atomAddF32(p + 3, v.w);
}

__global__ __launch_bounds__(256) void edge_scatter_bf16(
    float* __restrict__ agg, const __bf16* __restrict__ xin,
    const int* __restrict__ ei, int nEdges) {
  const int lane = threadIdx.x & 31;
  const int e    = blockIdx.x * 8 + (threadIdx.x >> 5);
  if (e >= nEdges) return;
  const int s = ei[e];
  const int d = ei[nEdges + e];
  v4bf v = *(const v4bf*)(xin + (size_t)s * 128 + lane * 4);
  float* p = agg + (size_t)d * 128 + lane * 4;
  atomAddF32(p + 0, (float)v[0]);
  atomAddF32(p + 1, (float)v[1]);
  atomAddF32(p + 2, (float)v[2]);
  atomAddF32(p + 3, (float)v[3]);
}

// ---- fused GIN MLP ---------------------------------------------------------
// stage bf16(x_in + agg) -> GEMM1 (BN+ReLU) -> GEMM2 (+b2 [+ReLU])
//   -> optional bf16 activation store (next layer input)
//   -> fused mean-pool accumulation into pooled[batch[row]] (f32, pre-round)
// 8 waves; wave w owns rows 16w..16w+15 end-to-end -> zero barriers needed.
__global__ __launch_bounds__(256) void gin_mlp(
    const float* __restrict__ agg,
    const float* __restrict__ xf,   // layer-0 input (f32)  -- or null
    const __bf16* __restrict__ xh,  // layer-1/2 input (bf16) -- or null
    const __bf16* __restrict__ wp1, const __bf16* __restrict__ wp2,
    const float* __restrict__ alpha, const float* __restrict__ beta,
    const float* __restrict__ b2,
    const int* __restrict__ batch, float* __restrict__ pooled,
    __bf16* __restrict__ hOut,      // bf16 activation out -- or null (layer 3)
    int colOff, int relu2, int nRows) {
  __shared__ __align__(16) __bf16 ldsT[128 * LDS_STRIDE];  // A tile, reused for H

  const int t       = threadIdx.x;
  const int lane    = t & 31;
  const int w       = t >> 5;
  const int rowBase = blockIdx.x * 128;

  __builtin_prefetch(wp1 + lane * 64, 0, 1);   // global_prefetch_b8, L2-resident
  __builtin_prefetch(wp2 + lane * 64, 0, 1);

  // ---- stage bf16(x_in + agg); each wave stages exactly its own 16 rows ----
  {
    const int r  = t >> 1;             // local row 0..127
    const int c0 = (t & 1) * 64;
    const int gr = rowBase + r;
    for (int c = 0; c < 64; c += 4) {
      float4 s = make_float4(0.f, 0.f, 0.f, 0.f);
      if (gr < nRows) {
        float4 a = *(const float4*)(agg + (size_t)gr * 128 + c0 + c);
        if (xf) {
          float4 v = *(const float4*)(xf + (size_t)gr * 128 + c0 + c);
          s = make_float4(a.x + v.x, a.y + v.y, a.z + v.z, a.w + v.w);
        } else {
          v4bf v = *(const v4bf*)(xh + (size_t)gr * 128 + c0 + c);
          s = make_float4(a.x + (float)v[0], a.y + (float)v[1],
                          a.z + (float)v[2], a.w + (float)v[3]);
        }
      }
      v4bf b;
      b[0] = (__bf16)s.x; b[1] = (__bf16)s.y; b[2] = (__bf16)s.z; b[3] = (__bf16)s.w;
      *(v4bf*)(&ldsT[r * LDS_STRIDE + c0 + c]) = b;
    }
  }

  const int waveRow = w * 16;
  const int colLane = lane & 15;
  const int rSub    = (lane & 16) ? 8 : 0;   // D layout: lanes 16-31 hold M=8..15

  // ---- GEMM1: H = relu(alpha*(A@W1)+beta), written back into ldsT ----------
  {
    v16bf a0 = load_afrag(ldsT, waveRow, lane, 0);
    v16bf a1 = load_afrag(ldsT, waveRow, lane, 1);
    v16bf a2 = load_afrag(ldsT, waveRow, lane, 2);
    v16bf a3 = load_afrag(ldsT, waveRow, lane, 3);
    for (int nt = 0; nt < 8; ++nt) {
      v8f acc = {0.f, 0.f, 0.f, 0.f, 0.f, 0.f, 0.f, 0.f};
      acc = wmma_bf16(a0, load_bfrag(wp1, lane, 0, nt), acc);
      acc = wmma_bf16(a1, load_bfrag(wp1, lane, 1, nt), acc);
      acc = wmma_bf16(a2, load_bfrag(wp1, lane, 2, nt), acc);
      acc = wmma_bf16(a3, load_bfrag(wp1, lane, 3, nt), acc);
      const int col = nt * 16 + colLane;
      const float al = alpha[col], be = beta[col];
#pragma unroll
      for (int i = 0; i < 8; ++i) {
        float h = acc[i] * al + be;
        h = h > 0.f ? h : 0.f;
        ldsT[(waveRow + rSub + i) * LDS_STRIDE + col] = (__bf16)h;
      }
    }
  }

  // rows this lane's D fragment touches (shared by all nt)
  int  gidx[8];
  bool valid[8];
#pragma unroll
  for (int i = 0; i < 8; ++i) {
    const int gr = rowBase + waveRow + rSub + i;
    valid[i] = (gr < nRows);
    gidx[i]  = valid[i] ? batch[gr] : 0;
  }

  // ---- GEMM2: out = H@W2 + b2 (+ReLU); bf16 store + fused pool-accumulate --
  {
    v16bf a0 = load_afrag(ldsT, waveRow, lane, 0);
    v16bf a1 = load_afrag(ldsT, waveRow, lane, 1);
    v16bf a2 = load_afrag(ldsT, waveRow, lane, 2);
    v16bf a3 = load_afrag(ldsT, waveRow, lane, 3);
    for (int nt = 0; nt < 8; ++nt) {
      v8f acc = {0.f, 0.f, 0.f, 0.f, 0.f, 0.f, 0.f, 0.f};
      acc = wmma_bf16(a0, load_bfrag(wp2, lane, 0, nt), acc);
      acc = wmma_bf16(a1, load_bfrag(wp2, lane, 1, nt), acc);
      acc = wmma_bf16(a2, load_bfrag(wp2, lane, 2, nt), acc);
      acc = wmma_bf16(a3, load_bfrag(wp2, lane, 3, nt), acc);
      const int col = nt * 16 + colLane;
      const float bb = b2[col];
#pragma unroll
      for (int i = 0; i < 8; ++i) {
        if (valid[i]) {
          const int gr = rowBase + waveRow + rSub + i;
          float o = acc[i] + bb;
          if (relu2) o = o > 0.f ? o : 0.f;
          if (hOut) hOut[(size_t)gr * 128 + col] = (__bf16)o;
          atomAddF32(&pooled[(size_t)gidx[i] * 384 + colOff + col], o);
        }
      }
    }
  }
}

// ---- per-graph node counts -------------------------------------------------
__global__ __launch_bounds__(256) void node_count(
    const int* __restrict__ batch, float* __restrict__ cnt, int nRows) {
  int n = blockIdx.x * 256 + threadIdx.x;
  if (n < nRows) atomAddF32(&cnt[batch[n]], 1.0f);
}

// ---- final linear: out[g] = (pooled[g]/cnt[g]) @ linW + linb ---------------
__global__ __launch_bounds__(128) void final_lin(
    const float* __restrict__ pooled, const float* __restrict__ cnt,
    const float* __restrict__ linW, const float* __restrict__ linb,
    float* __restrict__ out) {
  __shared__ float prow[384];
  const int g = blockIdx.x, c = threadIdx.x;
  const float inv = 1.0f / fmaxf(cnt[g], 1.0f);
  prow[c]       = pooled[(size_t)g * 384 + c] * inv;
  prow[c + 128] = pooled[(size_t)g * 384 + c + 128] * inv;
  prow[c + 256] = pooled[(size_t)g * 384 + c + 256] * inv;
  __syncthreads();
  float s = linb[c];
  for (int k = 0; k < 384; ++k) s = fmaf(prow[k], linW[k * 128 + c], s);
  out[(size_t)g * 128 + c] = s;
}

extern "C" void kernel_launch(void* const* d_in, const int* in_sizes, int n_in,
                              void* d_out, int out_size, void* d_ws, size_t ws_size,
                              hipStream_t stream) {
  const float* x     = (const float*)d_in[0];
  const int*   ei    = (const int*)d_in[1];
  const int*   batch = (const int*)d_in[2];
  const int N = in_sizes[0] / 128;
  const int E = in_sizes[1] / 2;

  const float* W1[3], *B1[3], *GM[3], *BT[3], *RM[3], *RV[3], *W2[3], *B2[3];
  for (int l = 0; l < 3; ++l) {
    int b = 3 + l * 8;
    W1[l] = (const float*)d_in[b + 0]; B1[l] = (const float*)d_in[b + 1];
    GM[l] = (const float*)d_in[b + 2]; BT[l] = (const float*)d_in[b + 3];
    RM[l] = (const float*)d_in[b + 4]; RV[l] = (const float*)d_in[b + 5];
    W2[l] = (const float*)d_in[b + 6]; B2[l] = (const float*)d_in[b + 7];
  }
  const float* linW = (const float*)d_in[27];
  const float* linb = (const float*)d_in[28];

  // workspace carve-up
  char* base = (char*)d_ws;
  float*  agg    = (float*)base;   base += (size_t)N * 128 * 4;
  __bf16* h0     = (__bf16*)base;  base += (size_t)N * 128 * 2;
  __bf16* h1     = (__bf16*)base;  base += (size_t)N * 128 * 2;
  __bf16* wpack  = (__bf16*)base;  base += (size_t)6 * 16384 * 2;
  float*  ab     = (float*)base;   base += (size_t)3 * 256 * 4;
  float*  pooled = (float*)base;   base += (size_t)GDIM * 384 * 4;
  float*  cnt    = (float*)base;

  // --- prep: pack weights, fold BN, zero pool accumulators ---
  pack_w<<<(6 * 16384 + 255) / 256, 256, 0, stream>>>(
      W1[0], W2[0], W1[1], W2[1], W1[2], W2[2], wpack);
  for (int l = 0; l < 3; ++l)
    fold_bn<<<1, 128, 0, stream>>>(B1[l], GM[l], BT[l], RM[l], RV[l],
                                   ab + l * 256, ab + l * 256 + 128);
  const int nz = GDIM * 384 + GDIM;
  zero_f32<<<(nz + 255) / 256, 256, 0, stream>>>(pooled, nz);
  node_count<<<(N + 255) / 256, 256, 0, stream>>>(batch, cnt, N);

  // --- 3 GIN layers ---
  const int nGemmBlk = (N + 127) / 128;
  const int nAgg     = N * 128;
  __bf16* hbuf[2] = {h0, h1};
  for (int l = 0; l < 3; ++l) {
    const float*  xf = (l == 0) ? x : nullptr;
    const __bf16* xh = (l == 0) ? nullptr : hbuf[(l - 1) & 1];
    __bf16* ho       = (l < 2) ? hbuf[l & 1] : nullptr;

    zero_f32<<<(nAgg + 255) / 256, 256, 0, stream>>>(agg, nAgg);
    if (xf) edge_scatter_f32 <<<(E + 7) / 8, 256, 0, stream>>>(agg, xf, ei, E);
    else    edge_scatter_bf16<<<(E + 7) / 8, 256, 0, stream>>>(agg, xh, ei, E);
    gin_mlp<<<nGemmBlk, 256, 0, stream>>>(
        agg, xf, xh,
        wpack + (size_t)(2 * l) * 16384, wpack + (size_t)(2 * l + 1) * 16384,
        ab + l * 256, ab + l * 256 + 128, B2[l],
        batch, pooled, ho, l * 128, (l < 2) ? 1 : 0, N);
  }

  // --- final linear ---
  final_lin<<<GDIM, 128, 0, stream>>>(pooled, cnt, linW, linb, (float*)d_out);
}
